// SimpleGCNLayer_15479062135162
// MI455X (gfx1250) — matrix operand order repaired and checked by reading, not compiled, not run on hardware
//
#include <hip/hip_runtime.h>
#include <hip/hip_bf16.h>
#include <stdint.h>

// ---------------------------------------------------------------------------
// SimpleGCNLayer for MI455X (gfx1250, wave32, WMMA)
//   out = diag(ds) * adj * diag(ds) * x * W^T + b,  ds = rsqrt(rowsum(adj)+eps)
// Folded as:  zT[j][k] = ds[k] * dot(x[k,:], W[j,:])   (bf16, 2MB, L2-resident)
//             out[i][j] = ds[i] * sum_k adj[i][k]*zT[j][k] + b[j]   (WMMA bf16)
// adj (256MB fp32) is streamed twice (rowsum + GEMM) -> ~22us HBM floor.
// adj traffic is non-temporal so the L2 keeps zT hot; the GEMM inner loop is
// software-pipelined one K-chunk ahead so waits never drain to loadcnt==0.
// ---------------------------------------------------------------------------

#define NN 8192
#define DD 128

typedef __attribute__((ext_vector_type(16))) __bf16   v16bf;
typedef __attribute__((ext_vector_type(8)))  float    v8f;
typedef __attribute__((ext_vector_type(8)))  uint32_t v8u;
typedef __attribute__((ext_vector_type(4)))  uint32_t v4u;
typedef __attribute__((ext_vector_type(4)))  float    v4f;

static __device__ __forceinline__ v4f nt_load4f(const float* p) {
#if __has_builtin(__builtin_nontemporal_load)
  return __builtin_nontemporal_load((const v4f*)p);
#else
  return *(const v4f*)p;
#endif
}

static __device__ __forceinline__ void nt_store1f(float* p, float v) {
#if __has_builtin(__builtin_nontemporal_store)
  __builtin_nontemporal_store(v, p);
#else
  *p = v;
#endif
}

// Pack two fp32 -> one dword of two bf16 (round-half-up via +0x8000, then
// v_perm_b32 grabs the two high halves in one VALU op).
static __device__ __forceinline__ uint32_t pack_bf16(float lo, float hi) {
  uint32_t ul = __builtin_bit_cast(uint32_t, lo) + 0x8000u;
  uint32_t uh = __builtin_bit_cast(uint32_t, hi) + 0x8000u;
#if __has_builtin(__builtin_amdgcn_perm)
  // result byte0,1 = ul bytes 2,3 ; byte2,3 = uh bytes 2,3
  return __builtin_amdgcn_perm(uh, ul, 0x07060302u);
#else
  return (ul >> 16) | (uh & 0xFFFF0000u);
#endif
}

static __device__ __forceinline__ uint16_t f32_to_bf16(float f) {
  uint32_t u = __builtin_bit_cast(uint32_t, f) + 0x8000u;
  return (uint16_t)(u >> 16);
}

// ---------------------------------------------------------------------------
// Kernel 1: ds[row] = rsqrt(rowsum(adj) + eps). One block per row, NT b128.
// ---------------------------------------------------------------------------
__global__ void __launch_bounds__(256) k_rowsum(const float* __restrict__ adj,
                                                float* __restrict__ ds) {
  __shared__ float red[256];
  const int row = blockIdx.x;
  const int t   = threadIdx.x;
  const float* ar = adj + (size_t)row * NN;
  float s = 0.f;
#pragma unroll
  for (int i = 0; i < 8; ++i) {            // 8 * 256 * 4 = 8192 floats
    v4f v = nt_load4f(ar + 4 * (t + i * 256));
    s += v.x + v.y + v.z + v.w;
  }
  red[t] = s;
  __syncthreads();
  for (int off = 128; off > 0; off >>= 1) {
    if (t < off) red[t] += red[t + off];
    __syncthreads();
  }
  if (t == 0) ds[row] = rsqrtf(red[0] + 1e-6f);
}

// ---------------------------------------------------------------------------
// Kernel 2: zT[j][k] = bf16( ds[k] * dot(x[k,:], W[j,:]) )   (transposed store
// so the GEMM's B fragments are contiguous along K). 128 threads = feature j,
// 32 k-values per block; x chunk staged (pre-scaled by ds) in LDS.
// ---------------------------------------------------------------------------
__global__ void __launch_bounds__(128) k_zt(const float* __restrict__ x,
                                            const float* __restrict__ W,
                                            const float* __restrict__ ds,
                                            uint16_t* __restrict__ zT) {
  __shared__ float xs[32 * 128];           // 16 KB
  const int t     = threadIdx.x;           // 0..127 (= output feature j)
  const int kbase = blockIdx.x * 32;

  const v4f* xg  = (const v4f*)(x + (size_t)kbase * DD);
  v4f*       xls = (v4f*)xs;
#pragma unroll
  for (int i = 0; i < 8; ++i) {
    int idx4 = t + i * 128;                // 0..1023 float4s
    int kl   = idx4 >> 5;                  // local k (32 float4 per row)
    v4f v = xg[idx4];
    v = v * ds[kbase + kl];
    xls[idx4] = v;
  }
  __syncthreads();

  const v4f* wr = (const v4f*)(W + (size_t)t * DD);
  v4f wreg[32];
#pragma unroll
  for (int c = 0; c < 32; ++c) wreg[c] = wr[c];

  for (int kk = 0; kk < 32; ++kk) {
    const v4f* xv = (const v4f*)(xs + kk * DD);
    float s = 0.f;
#pragma unroll
    for (int c = 0; c < 32; ++c) {
      v4f a = wreg[c], b2 = xv[c];
      s += a.x * b2.x + a.y * b2.y + a.z * b2.z + a.w * b2.w;
    }
    zT[(size_t)t * NN + (kbase + kk)] = f32_to_bf16(s);
  }
}

// ---------------------------------------------------------------------------
// Kernel 3: the big GEMM via v_wmma_f32_16x16x32_bf16, software-pipelined.
// Grid 256 blocks x 256 threads (8 waves). Block = 32 output rows x 128 cols.
// Wave w: row tile (w&1), col pair (w>>1) -> two 16x16 tiles, 2 WMMA / chunk.
// A frag (16x32 bf16): lane = row M (lane&15), hi = lane>>4;
//   elems 0..7  = adj[M][kb + 8*hi + 0..7]
//   elems 8..15 = adj[M][kb + 16 + 8*hi + 0..7]      (per ISA VGPR layout)
// B frag (32x16 bf16): lane = col N; same K order as A (8*hi + 0..7,
//   16 + 8*hi + 0..7) in zT[col][...] -> two 16B loads, no shuffling.
// Loads for chunk kb+32 are issued into fresh registers before the WMMAs for
// chunk kb, so the pre-WMMA wait only has to cover a full iteration-old group.
// ---------------------------------------------------------------------------
struct Q2 { v4u a, b; };                   // 32-byte carrier for a B fragment

__global__ void __launch_bounds__(256) k_gcn_wmma(const float*    __restrict__ adj,
                                                  const uint16_t* __restrict__ zT,
                                                  const float*    __restrict__ ds,
                                                  const float*    __restrict__ bias,
                                                  float*          __restrict__ out) {
  const int tid   = threadIdx.x;
  const int wave  = tid >> 5;
  const int lane  = tid & 31;
  const int laneM = lane & 15;
  const int hi    = lane >> 4;

  const int rowTile = wave & 1;            // 0..1
  const int colPair = wave >> 1;           // 0..3
  const int rowBase = blockIdx.x * 32 + rowTile * 16;
  const int colBase = colPair * 32;

  const float*    aptr  = adj + (size_t)(rowBase + laneM) * NN + 8 * hi;
  const uint16_t* zptr0 = zT + (size_t)(colBase + laneM)      * NN + 8 * hi;
  const uint16_t* zptr1 = zT + (size_t)(colBase + 16 + laneM) * NN + 8 * hi;

  v8f acc0 = {};
  v8f acc1 = {};

  // ---- prologue: loads for chunk kb = 0 ----
  v4f a0 = nt_load4f(aptr);
  v4f a1 = nt_load4f(aptr + 4);
  v4f a2 = nt_load4f(aptr + 16);
  v4f a3 = nt_load4f(aptr + 20);
  Q2 q0, q1;
  { const v4u* p = (const v4u*)zptr0; q0.a = p[0]; q0.b = p[2]; }
  { const v4u* p = (const v4u*)zptr1; q1.a = p[0]; q1.b = p[2]; }

  for (int kb = 0; kb < NN; kb += 32) {
    // next chunk (clamped in-bounds; final iteration harmlessly reloads 0)
    const int kn = (kb + 32 < NN) ? (kb + 32) : 0;

    // keep the adj HBM stream ~4KB ahead (speculative; dropped if unmapped)
    __builtin_prefetch(aptr + kb + 1024, 0, 1);     // -> global_prefetch_b8

    v4f na0 = nt_load4f(aptr + kn);
    v4f na1 = nt_load4f(aptr + kn + 4);
    v4f na2 = nt_load4f(aptr + kn + 16);
    v4f na3 = nt_load4f(aptr + kn + 20);
    Q2 nq0, nq1;
    { const v4u* p = (const v4u*)(zptr0 + kn); nq0.a = p[0]; nq0.b = p[2]; }
    { const v4u* p = (const v4u*)(zptr1 + kn); nq1.a = p[0]; nq1.b = p[2]; }

    // ---- compute on current chunk ----
    v8u au;
    au[0] = pack_bf16(a0.x, a0.y); au[1] = pack_bf16(a0.z, a0.w);
    au[2] = pack_bf16(a1.x, a1.y); au[3] = pack_bf16(a1.z, a1.w);
    au[4] = pack_bf16(a2.x, a2.y); au[5] = pack_bf16(a2.z, a2.w);
    au[6] = pack_bf16(a3.x, a3.y); au[7] = pack_bf16(a3.z, a3.w);
    v16bf afrag  = __builtin_bit_cast(v16bf, au);
    v16bf bfrag0 = __builtin_bit_cast(v16bf, q0);
    v16bf bfrag1 = __builtin_bit_cast(v16bf, q1);

    acc0 = __builtin_amdgcn_wmma_f32_16x16x32_bf16(
        false, afrag, false, bfrag0, (short)0, acc0, false, false);
    acc1 = __builtin_amdgcn_wmma_f32_16x16x32_bf16(
        false, afrag, false, bfrag1, (short)0, acc1, false, false);

    a0 = na0; a1 = na1; a2 = na2; a3 = na3;
    q0 = nq0; q1 = nq1;
  }

  // ---- epilogue: out[m][n] = ds[m]*acc + b[n] ----
  // D layout: VGPR r <-> M = r + 8*hi ; lane (0..15) <-> N.
  const float bv0 = bias[colBase + laneM];
  const float bv1 = bias[colBase + 16 + laneM];
#pragma unroll
  for (int r = 0; r < 8; ++r) {
    const int m   = rowBase + r + 8 * hi;
    const float d = ds[m];
    nt_store1f(out + (size_t)m * DD + colBase + laneM,      d * acc0[r] + bv0);
    nt_store1f(out + (size_t)m * DD + colBase + 16 + laneM, d * acc1[r] + bv1);
  }
}

// ---------------------------------------------------------------------------
extern "C" void kernel_launch(void* const* d_in, const int* in_sizes, int n_in,
                              void* d_out, int out_size, void* d_ws, size_t ws_size,
                              hipStream_t stream) {
  const float* x   = (const float*)d_in[0];   // [8192,128]
  const float* adj = (const float*)d_in[1];   // [8192,8192]
  const float* W   = (const float*)d_in[2];   // [128,128]
  const float* b   = (const float*)d_in[3];   // [128]
  float* out = (float*)d_out;                 // [8192,128]

  // workspace: ds (32KB fp32) | zT (128 x 8192 bf16 = 2MB)
  float*    ds = (float*)d_ws;
  uint16_t* zT = (uint16_t*)((char*)d_ws + NN * sizeof(float));

  k_rowsum  <<<NN,      256, 0, stream>>>(adj, ds);
  k_zt      <<<NN / 32, 128, 0, stream>>>(x, W, ds, zT);
  k_gcn_wmma<<<NN / 32, 256, 0, stream>>>(adj, zT, ds, b, out);
}
